// FlowRefineNetA_28097676051234
// MI455X (gfx1250) — matrix operand order repaired and checked by reading, not compiled, not run on hardware
//
#include <hip/hip_runtime.h>
#include <hip/hip_bf16.h>
#include <math.h>

// ---------------------------------------------------------------------------
// FlowRefineNet on MI455X (gfx1250, wave32).
//  * fused warp/unfold/occl-MLP/l2norm/correlation gather kernel (L2-resident)
//  * concatenated conv inputs materialized as contiguous channel containers
//    (producers write straight into their slice -> single-tensor conv reads)
//  * conv layers as fully-specialized WMMA GEMMs: v_wmma_f32_16x16x32_f16,
//    M=16 out-ch, N=16 pixels/row-tile, K=32 chunks of flattened (cin,tap).
//    A fragments pre-packed to f16 in LDS in contiguous k = 16*half + s order.
// ---------------------------------------------------------------------------

#define BB   2
#define HH   192
#define WW   192
#define HWSZ (HH * WW)
#define CTXC 64
#define GINC 146   // gru input container channels: fea(64)|feaX(64)|m(16)|f(2)

typedef __attribute__((ext_vector_type(16))) _Float16 v16h;
typedef __attribute__((ext_vector_type(8)))  float    v8f;

// ------------------------------ fused gather ------------------------------

struct Samp { int x0i, y0i, x1i, y1i; float wx, wy; };

__device__ __forceinline__ Samp mk_samp(float fx, float fy, int x, int y) {
  float gx = fminf(fmaxf((float)x + fx, 0.f), (float)(WW - 1));
  float gy = fminf(fmaxf((float)y + fy, 0.f), (float)(HH - 1));
  float fx0 = floorf(gx), fy0 = floorf(gy);
  Samp s;
  s.wx  = gx - fx0;  s.wy  = gy - fy0;
  s.x0i = (int)fx0;  s.y0i = (int)fy0;
  s.x1i = min(s.x0i + 1, WW - 1);
  s.y1i = min(s.y0i + 1, HH - 1);
  return s;
}

// sample of warp(unfold3(img))[c, p=(i,j)] at this pixel; zero-padded borders
__device__ __forceinline__ float samp(const float* __restrict__ img, int b, int c,
                                      const Samp& s, int i, int j) {
  const float* p = img + ((size_t)(b * CTXC + c)) * HWSZ;
  int ya = s.y0i + i - 1, yb = s.y1i + i - 1;
  int xa = s.x0i + j - 1, xb = s.x1i + j - 1;
  bool yaok = (ya >= 0) & (ya < HH), ybok = (yb >= 0) & (yb < HH);
  bool xaok = (xa >= 0) & (xa < WW), xbok = (xb >= 0) & (xb < WW);
  float v00 = (yaok && xaok) ? p[ya * WW + xa] : 0.f;
  float v01 = (yaok && xbok) ? p[ya * WW + xb] : 0.f;
  float v10 = (ybok && xaok) ? p[yb * WW + xa] : 0.f;
  float v11 = (ybok && xbok) ? p[yb * WW + xb] : 0.f;
  return (1.f - s.wy) * ((1.f - s.wx) * v00 + s.wx * v01) +
         s.wy        * ((1.f - s.wx) * v10 + s.wx * v11);
}

__global__ void corr_fused_kernel(
    const float* __restrict__ x0, const float* __restrict__ x1,
    const float* __restrict__ f0, const float* __restrict__ f1,
    const float* __restrict__ w1, const float* __restrict__ b1, const float* __restrict__ a1,
    const float* __restrict__ w2, const float* __restrict__ b2, const float* __restrict__ a2,
    const float* __restrict__ w3, const float* __restrict__ b3,
    float* __restrict__ gin0, float* __restrict__ gin1,
    float* __restrict__ corr0O, float* __restrict__ corr1O)
{
  extern __shared__ float smem[];
  float* sW1 = smem;         // 16 x 128
  float* sB1 = sW1 + 2048;   // 16
  float* sA1 = sB1 + 16;     // 16
  float* sW2 = sA1 + 16;     // 16 x 16
  float* sB2 = sW2 + 256;    // 16
  float* sA2 = sB2 + 16;     // 16
  float* sW3 = sA2 + 16;     // 16
  for (int i = threadIdx.x; i < 2048; i += blockDim.x) sW1[i] = w1[i];
  for (int i = threadIdx.x; i < 256;  i += blockDim.x) sW2[i] = w2[i];
  if (threadIdx.x < 16) {
    sB1[threadIdx.x] = b1[threadIdx.x];
    sA1[threadIdx.x] = a1[threadIdx.x];
    sB2[threadIdx.x] = b2[threadIdx.x];
    sA2[threadIdx.x] = a2[threadIdx.x];
    sW3[threadIdx.x] = w3[threadIdx.x];
  }
  __syncthreads();

  int idx = blockIdx.x * blockDim.x + threadIdx.x;
  const int total = BB * HWSZ;
  if (idx >= total) return;
  int x  = idx % WW;
  int y  = (idx / WW) % HH;
  int b  = idx / HWSZ;
  size_t pixoff = (size_t)y * WW + x;

  const size_t fbase = (size_t)b * 2 * HWSZ + pixoff;
  Samp s0 = mk_samp(f0[fbase], f0[fbase + HWSZ], x, y);
  Samp s1 = mk_samp(f1[fbase], f1[fbase + HWSZ], x, y);

  // container slice bases (channel offsets inside gin: fea@0, feaX@64)
  const size_t gFea  = (size_t)(b * GINC) * HWSZ + pixoff;        // + c*HWSZ
  const size_t gFeaX = (size_t)(b * GINC + 64) * HWSZ + pixoff;   // + c*HWSZ

  // ---- pass A: center features + occlusion-MLP hidden accumulation
  float h1[16];
  #pragma unroll
  for (int k = 0; k < 16; ++k) h1[k] = sB1[k];
  for (int c = 0; c < CTXC; ++c) {
    float v0 = samp(x0, b, c, s0, 1, 1);
    float v1 = samp(x1, b, c, s1, 1, 1);
    gin0[gFeaX + (size_t)c * HWSZ] = v0;
    gin1[gFeaX + (size_t)c * HWSZ] = v1;
    #pragma unroll
    for (int k = 0; k < 16; ++k)
      h1[k] += sW1[k * 128 + c] * v0 + sW1[k * 128 + 64 + c] * v1;
  }
  #pragma unroll
  for (int k = 0; k < 16; ++k) h1[k] = (h1[k] >= 0.f) ? h1[k] : sA1[k] * h1[k];
  float h2[16];
  #pragma unroll
  for (int k = 0; k < 16; ++k) {
    float acc = sB2[k];
    #pragma unroll
    for (int j = 0; j < 16; ++j) acc += sW2[k * 16 + j] * h1[j];
    h2[k] = (acc >= 0.f) ? acc : sA2[k] * acc;
  }
  float o = b3[0];
  #pragma unroll
  for (int j = 0; j < 16; ++j) o += sW3[j] * h2[j];
  o = 1.f / (1.f + __expf(-o));

  // ---- pass B: blended feature (into both containers) + squared norm
  float ss = 1e-5f;
  for (int c = 0; c < CTXC; ++c) {
    size_t co = (size_t)c * HWSZ;
    float f = gin0[gFeaX + co] * o + gin1[gFeaX + co] * (1.f - o);
    gin0[gFea + co] = f;
    gin1[gFea + co] = f;
    ss += f * f;
  }
  float invn = rsqrtf(ss);

  // ---- pass C: 9-point normalized correlation, both images
  for (int side = 0; side < 2; ++side) {
    const float* img = side ? x1 : x0;
    Samp sp = side ? s1 : s0;
    float* corrO = side ? corr1O : corr0O;
    for (int p = 0; p < 9; ++p) {
      int pi = p / 3, pj = p % 3;
      float dot = 0.f, nrm = 1e-5f;
      for (int c = 0; c < CTXC; ++c) {
        float v  = samp(img, b, c, sp, pi, pj);
        float fv = gin0[gFea + (size_t)c * HWSZ];
        dot += fv * v;
        nrm += v * v;
      }
      corrO[((size_t)(b * 9 + p)) * HWSZ + pixoff] = dot * invn * rsqrtf(nrm);
    }
  }
}

// ------------------------------ WMMA conv ---------------------------------

struct ConvP {
  const float* in;     // single planar (B, CIN, H, W) tensor
  const float* w;      // (COUT, CIN, k, k)
  const float* bias;   // (COUT)
  const float* alpha;  // PReLU alpha or null
  const float* addto;  // residual add (flow update) or null
  float* out;          // (B, CTOT, H, W), written at channel offset COFF
};

template <int CIN, int KSZ, int COUT, int CTOT, int COFF, bool PRELU, bool ADD>
__global__ void conv_wmma(ConvP p) {
  constexpr int TT     = KSZ * KSZ;
  constexpr int R      = CIN * TT;
  constexpr int NCHUNK = (R + 31) / 32;
  constexpr int PAD    = KSZ >> 1;

  extern __shared__ __align__(32) char smraw[];
  v16h*  aPack = (v16h*)smraw;                               // NCHUNK*32 frags
  float* sb    = (float*)(smraw + (size_t)NCHUNK * 32 * sizeof(v16h));
  float* sa    = sb + 16;

  // ---- cooperative pre-pack of A fragments (f16, contiguous k = 16*hi + h)
  for (int e = threadIdx.x; e < NCHUNK * 32; e += blockDim.x) {
    int chunk = e >> 5, ln = e & 31;
    int lm = ln & 15, hi = ln >> 4;
    v16h v;
    #pragma unroll
    for (int h = 0; h < 16; ++h) {
      int Kidx = (hi << 4) + h;
      int r = chunk * 32 + Kidx;
      float wv = 0.f;
      if (r < R && lm < COUT) {
        int cin = r / TT, tap = r % TT;
        wv = p.w[(lm * CIN + cin) * TT + tap];
      }
      v[h] = (_Float16)wv;
    }
    aPack[e] = v;
  }
  if (threadIdx.x < 16) {
    int i = threadIdx.x;
    sb[i] = (i < COUT) ? p.bias[i] : 0.f;
    sa[i] = (PRELU && i < COUT) ? p.alpha[i] : 0.f;
  }
  __syncthreads();

  const int nTiles = BB * HH * (WW / 16);
  int wave = blockIdx.x * (blockDim.x >> 5) + (threadIdx.x >> 5);
  if (wave >= nTiles) return;
  int lane  = threadIdx.x & 31;
  int laneM = lane & 15;

  int x0p = (wave % (WW / 16)) * 16;
  int y   = (wave / (WW / 16)) % HH;
  int b   = wave / ((WW / 16) * HH);

  int  yy[KSZ];  bool yok[KSZ];
  int  xxv[KSZ]; bool xok[KSZ];
  #pragma unroll
  for (int t = 0; t < KSZ; ++t) {
    yy[t]  = y + t - PAD;           yok[t] = (yy[t] >= 0) && (yy[t] < HH);
    xxv[t] = x0p + laneM + t - PAD; xok[t] = (xxv[t] >= 0) && (xxv[t] < WW);
  }
  const float* inB = p.in + (size_t)(b * CIN) * HWSZ;

  v8f acc = {0.f, 0.f, 0.f, 0.f, 0.f, 0.f, 0.f, 0.f};

  #pragma clang loop unroll(full)
  for (int chunk = 0; chunk < NCHUNK; ++chunk) {
    v16h afrag = aPack[(chunk << 5) + lane];
    v16h bfrag;
    // physical K slot = (lane>=16 ? 16 : 0) + h; split so logical r is a
    // compile-time constant in each half (full reconvergence before WMMA)
    if (lane < 16) {
      #pragma unroll
      for (int h = 0; h < 16; ++h) {
        int r = chunk * 32 + h;
        float v = 0.f;
        if (r < R) {
          int cin = r / TT, tap = r % TT, ky = tap / KSZ, kx = tap % KSZ;
          if (yok[ky] && xok[kx])
            v = inB[(size_t)cin * HWSZ + yy[ky] * WW + xxv[kx]];
        }
        bfrag[h] = (_Float16)v;
      }
    } else {
      #pragma unroll
      for (int h = 0; h < 16; ++h) {
        int r = chunk * 32 + 16 + h;
        float v = 0.f;
        if (r < R) {
          int cin = r / TT, tap = r % TT, ky = tap / KSZ, kx = tap % KSZ;
          if (yok[ky] && xok[kx])
            v = inB[(size_t)cin * HWSZ + yy[ky] * WW + xxv[kx]];
        }
        bfrag[h] = (_Float16)v;
      }
    }
    acc = __builtin_amdgcn_wmma_f32_16x16x32_f16(
        false, afrag, false, bfrag, (short)0, acc, false, false);
  }

  // ---- epilogue: bias, PReLU, optional residual add, container placement
  int pix = x0p + laneM;
  int hi  = lane >> 4;
  #pragma unroll
  for (int r8 = 0; r8 < 8; ++r8) {
    int m = (hi << 3) + r8;
    if (m < COUT) {
      float v = acc[r8] + sb[m];
      if (PRELU) v = (v >= 0.f) ? v : sa[m] * v;
      size_t oidx = ((size_t)(b * CTOT + COFF + m) * HH + y) * WW + pix;
      if (ADD) v += p.addto[((size_t)(b * COUT + m) * HH + y) * WW + pix];
      p.out[oidx] = v;
    }
  }
}

// --------------------- flow -> gru-container copy -------------------------

__global__ void copy_flow_kernel(const float* __restrict__ fc0,
                                 const float* __restrict__ fc1,
                                 float* __restrict__ gin0,
                                 float* __restrict__ gin1) {
  int idx = blockIdx.x * blockDim.x + threadIdx.x;
  const int total = BB * 2 * HWSZ;   // per side
  if (idx >= total) return;
  int pix = idx % HWSZ;
  int c   = (idx / HWSZ) % 2;
  int b   = idx / (2 * HWSZ);
  size_t dst = ((size_t)(b * GINC + 144 + c)) * HWSZ + pix;
  gin0[dst] = fc0[idx];
  gin1[dst] = fc1[idx];
}

// ------------------------------ output concat -----------------------------

__global__ void concat_out_kernel(const float* __restrict__ f0,
                                  const float* __restrict__ f1,
                                  float* __restrict__ out) {
  int idx = blockIdx.x * blockDim.x + threadIdx.x;
  const int total = BB * 4 * HWSZ;
  if (idx >= total) return;
  int pix = idx % HWSZ;
  int c   = (idx / HWSZ) % 4;
  int b   = idx / (4 * HWSZ);
  float v = (c < 2) ? f0[((size_t)(b * 2 + c)) * HWSZ + pix]
                    : f1[((size_t)(b * 2 + (c - 2))) * HWSZ + pix];
  out[idx] = v;
}

// ------------------------------ host side ---------------------------------

template <int CIN, int KSZ, int COUT, int CTOT, int COFF, bool PRELU, bool ADD>
static void launch_conv_t(hipStream_t st, ConvP p) {
  constexpr int R      = CIN * KSZ * KSZ;
  constexpr int NCHUNK = (R + 31) / 32;
  const int nTiles = BB * HH * (WW / 16);       // 4608 row-tiles of 16 pixels
  int blocks = (nTiles + 7) / 8;                // 8 waves / block
  size_t lds = (size_t)NCHUNK * 32 * sizeof(v16h) + 32 * sizeof(float);
  conv_wmma<CIN, KSZ, COUT, CTOT, COFF, PRELU, ADD><<<blocks, 256, lds, st>>>(p);
}

extern "C" void kernel_launch(void* const* d_in, const int* in_sizes, int n_in,
                              void* d_out, int out_size, void* d_ws, size_t ws_size,
                              hipStream_t stream) {
  const float* x0       = (const float*)d_in[0];
  const float* x1       = (const float*)d_in[1];
  const float* flow0_in = (const float*)d_in[2];
  const float* flow1_in = (const float*)d_in[3];
  // params flattened: top dict in insertion order, nested pytree with sorted
  // dict keys (corr, flow, gru, head, motion, occl; conv_p = {b, w}).
  int i = 4;
  const float* corr0b = (const float*)d_in[i++];
  const float* corr0w = (const float*)d_in[i++];
  const float* corrA0 = (const float*)d_in[i++];
  const float* corr1b = (const float*)d_in[i++];
  const float* corr1w = (const float*)d_in[i++];
  const float* corrA1 = (const float*)d_in[i++];
  const float* flw0b  = (const float*)d_in[i++];
  const float* flw0w  = (const float*)d_in[i++];
  const float* flwA0  = (const float*)d_in[i++];
  const float* flw1b  = (const float*)d_in[i++];
  const float* flw1w  = (const float*)d_in[i++];
  const float* flwA1  = (const float*)d_in[i++];
  const float* gru0b  = (const float*)d_in[i++];
  const float* gru0w  = (const float*)d_in[i++];
  const float* gruA0  = (const float*)d_in[i++];
  const float* gru1b  = (const float*)d_in[i++];
  const float* gru1w  = (const float*)d_in[i++];
  const float* gruA1  = (const float*)d_in[i++];
  const float* head0b = (const float*)d_in[i++];
  const float* head0w = (const float*)d_in[i++];
  const float* headA0 = (const float*)d_in[i++];
  const float* head1b = (const float*)d_in[i++];
  const float* head1w = (const float*)d_in[i++];
  const float* motb   = (const float*)d_in[i++];
  const float* motw   = (const float*)d_in[i++];
  const float* motA   = (const float*)d_in[i++];
  const float* occl0b = (const float*)d_in[i++];
  const float* occl0w = (const float*)d_in[i++];
  const float* occlA0 = (const float*)d_in[i++];
  const float* occl1b = (const float*)d_in[i++];
  const float* occl1w = (const float*)d_in[i++];
  const float* occlA1 = (const float*)d_in[i++];
  const float* occl2b = (const float*)d_in[i++];
  const float* occl2w = (const float*)d_in[i++];

  // workspace layout (floats)
  float* wsp = (float*)d_ws;
  size_t o = 0;
  float* gin0   = wsp + o; o += (size_t)BB * GINC * HWSZ;  // fea|fea0|m0|f0
  float* gin1   = wsp + o; o += (size_t)BB * GINC * HWSZ;  // fea|fea1|m1|f1
  float* corr0W = wsp + o; o += (size_t)BB * 9 * HWSZ;
  float* corr1W = wsp + o; o += (size_t)BB * 9 * HWSZ;
  float* mcat   = wsp + o; o += (size_t)BB * 32 * HWSZ;    // corrf(16)|ff(16)
  float* tA     = wsp + o; o += (size_t)BB * 16 * HWSZ;
  float* tB     = wsp + o; o += (size_t)BB * 16 * HWSZ;
  float* f0A    = wsp + o; o += (size_t)BB * 2 * HWSZ;
  float* f0B    = wsp + o; o += (size_t)BB * 2 * HWSZ;
  float* f1A    = wsp + o; o += (size_t)BB * 2 * HWSZ;
  float* f1B    = wsp + o; o += (size_t)BB * 2 * HWSZ;

  const float* fc0 = flow0_in;
  const float* fc1 = flow1_in;

  for (int it = 0; it < 4; ++it) {
    float* fn0 = (it & 1) ? f0B : f0A;
    float* fn1 = (it & 1) ? f1B : f1A;

    corr_fused_kernel<<<(BB * HWSZ + 255) / 256, 256, 2384 * sizeof(float), stream>>>(
        x0, x1, fc0, fc1,
        occl0w, occl0b, occlA0, occl1w, occl1b, occlA1, occl2w, occl2b,
        gin0, gin1, corr0W, corr1W);

    copy_flow_kernel<<<(BB * 2 * HWSZ + 255) / 256, 256, 0, stream>>>(
        fc0, fc1, gin0, gin1);

    for (int s = 0; s < 2; ++s) {
      const float* corrS = s ? corr1W : corr0W;
      const float* fS    = s ? fc1 : fc0;
      float* ginS        = s ? gin1 : gin0;
      float* fOut        = s ? fn1 : fn0;

      // corr path: 1x1 9->16 (tA), 1x1 16->16 into mcat[0:16]
      launch_conv_t<9, 1, 16, 16, 0, true, false>(stream,
          {corrS, corr0w, corr0b, corrA0, nullptr, tA});
      launch_conv_t<16, 1, 16, 32, 0, true, false>(stream,
          {tA, corr1w, corr1b, corrA1, nullptr, mcat});
      // flow path: 3x3 2->16 (tA), 3x3 16->16 into mcat[16:32]
      launch_conv_t<2, 3, 16, 16, 0, true, false>(stream,
          {fS, flw0w, flw0b, flwA0, nullptr, tA});
      launch_conv_t<16, 3, 16, 32, 16, true, false>(stream,
          {tA, flw1w, flw1b, flwA1, nullptr, mcat});
      // motion: 3x3 32->16 into gin[128:144]
      launch_conv_t<32, 3, 16, GINC, 128, true, false>(stream,
          {mcat, motw, motb, motA, nullptr, ginS});
      // gru: 3x3 146->16 (tA), 3x3 16->16 (tB)
      launch_conv_t<GINC, 3, 16, 16, 0, true, false>(stream,
          {ginS, gru0w, gru0b, gruA0, nullptr, tA});
      launch_conv_t<16, 3, 16, 16, 0, true, false>(stream,
          {tA, gru1w, gru1b, gruA1, nullptr, tB});
      // head: 3x3 16->16 (prelu), 3x3 16->2 (+ residual flow)
      launch_conv_t<16, 3, 16, 16, 0, true, false>(stream,
          {tB, head0w, head0b, headA0, nullptr, tA});
      launch_conv_t<16, 3, 2, 2, 0, false, true>(stream,
          {tA, head1w, head1b, nullptr, fS, fOut});
    }
    fc0 = fn0;
    fc1 = fn1;
  }

  concat_out_kernel<<<(BB * 4 * HWSZ + 255) / 256, 256, 0, stream>>>(
      fc0, fc1, (float*)d_out);
}
